// MambaBlock_75900662055669
// MI455X (gfx1250) — matrix operand order repaired and checked by reading, not compiled, not run on hardware
//
#include <hip/hip_runtime.h>
#include <hip/hip_bf16.h>
#include <math.h>

// ---------------------------------------------------------------------------
// Mamba block for MI455X (gfx1250), wave32, WMMA bf16 path.
// GEMMs: LDS-staged 64x128 block tiles, 32x32 wave tiles (2x2 WMMA),
// v_wmma_f32_16x16x32_bf16 with fp32 accumulation.
// ---------------------------------------------------------------------------

typedef __attribute__((ext_vector_type(16))) __bf16 v16bf_t;
typedef __attribute__((ext_vector_type(8)))  __bf16 v8bf_t;
typedef __attribute__((ext_vector_type(8)))  float  v8f_t;

union FragBF { v16bf_t v; v8bf_t h[2]; };

static constexpr int Bb    = 2;
static constexpr int Ll    = 1024;
static constexpr int DM    = 1024;   // d_model
static constexpr int DI    = 2048;   // d_inner
static constexpr int DS    = 16;     // d_state
static constexpr int Mrows = Bb * Ll;   // 2048 GEMM rows

// GEMM tiling
static constexpr int BM = 64;    // block rows
static constexpr int BN = 128;   // block cols
static constexpr int BK = 32;    // k-step (WMMA K)
static constexpr int SA = 48;    // LDS row stride (elems): 96B, 16B-aligned, low-conflict
static constexpr int SB = 48;

// --------------------------- fp32 -> bf16 convert --------------------------
__global__ __launch_bounds__(256) void cvt_bf16_kernel(
    const float* __restrict__ in, __bf16* __restrict__ out, int n)
{
    int i = blockIdx.x * 256 + threadIdx.x;
    if (i < n) out[i] = (__bf16)in[i];
}

// --------------------------- input LayerNorm (writes bf16) -----------------
__global__ __launch_bounds__(256) void layernorm_in_kernel(
    const float* __restrict__ x, const float* __restrict__ alpha,
    const float* __restrict__ beta, __bf16* __restrict__ out)
{
    __shared__ float r1[256];
    __shared__ float r2[256];
    const int row = blockIdx.x;
    const float* xr = x + (size_t)row * DM;
    float s = 0.f, ss = 0.f;
    for (int c = threadIdx.x; c < DM; c += 256) {
        float v = xr[c];
        s += v; ss += v * v;
    }
    r1[threadIdx.x] = s; r2[threadIdx.x] = ss;
    __syncthreads();
    for (int k = 128; k > 0; k >>= 1) {
        if (threadIdx.x < k) {
            r1[threadIdx.x] += r1[threadIdx.x + k];
            r2[threadIdx.x] += r2[threadIdx.x + k];
        }
        __syncthreads();
    }
    float mean = r1[0] / (float)DM;
    float var  = fmaxf((r2[0] - (float)DM * mean * mean) / (float)(DM - 1), 0.f);
    float inv  = 1.f / (sqrtf(var) + 1e-6f);    // (x-mean)/(std+eps), ddof=1
    for (int c = threadIdx.x; c < DM; c += 256) {
        float v = alpha[c] * (xr[c] - mean) * inv + beta[c];
        out[(size_t)row * DM + c] = (__bf16)v;
    }
}

// --------------------------- final residual LayerNorm ----------------------
__global__ __launch_bounds__(256) void layernorm_out_kernel(
    const float* __restrict__ resid, const float* __restrict__ proj,
    const float* __restrict__ alpha, const float* __restrict__ beta,
    float* __restrict__ out)
{
    __shared__ float r1[256];
    __shared__ float r2[256];
    const int row = blockIdx.x;
    const float* xr = resid + (size_t)row * DM;
    const float* pr = proj  + (size_t)row * DM;
    float s = 0.f, ss = 0.f;
    for (int c = threadIdx.x; c < DM; c += 256) {
        float v = xr[c] + pr[c];
        s += v; ss += v * v;
    }
    r1[threadIdx.x] = s; r2[threadIdx.x] = ss;
    __syncthreads();
    for (int k = 128; k > 0; k >>= 1) {
        if (threadIdx.x < k) {
            r1[threadIdx.x] += r1[threadIdx.x + k];
            r2[threadIdx.x] += r2[threadIdx.x + k];
        }
        __syncthreads();
    }
    float mean = r1[0] / (float)DM;
    float var  = fmaxf((r2[0] - (float)DM * mean * mean) / (float)(DM - 1), 0.f);
    float inv  = 1.f / (sqrtf(var) + 1e-6f);
    for (int c = threadIdx.x; c < DM; c += 256) {
        float v = xr[c] + pr[c];
        out[(size_t)row * DM + c] = alpha[c] * (v - mean) * inv + beta[c];
    }
}

// --------------------------- WMMA bf16 GEMM (LDS-staged) -------------------
// C[M,N] = A[M,K] * W[N,K]^T + bias[N]   (all row-major)
// grid = (N/BN, M/BM), block = 256 (8 waves in 2x4), wave tile 32x32 (2x2 WMMA)
// mode 0: bias only; mode 1: bias + softplus + clip(1e-4, 10)  (dt_proj)
__global__ __launch_bounds__(256) void gemm_wmma_bf16_kernel(
    const __bf16* __restrict__ A, const __bf16* __restrict__ W,
    const float* __restrict__ bias, float* __restrict__ C,
    int M, int N, int K, int mode)
{
    __shared__ __attribute__((aligned(16))) __bf16 As[BM * SA];  //  6 KB
    __shared__ __attribute__((aligned(16))) __bf16 Bs[BN * SB];  // 12 KB

    const int tid  = threadIdx.x;
    const int lane = tid & 31;
    const int wave = tid >> 5;            // 0..7
    const int wm   = wave >> 2;           // 0..1  (wave row)
    const int wn   = wave & 3;            // 0..3  (wave col)
    const int l16  = lane & 15;
    const int half = lane >> 4;           // 0 => lanes 0-15, 1 => lanes 16-31

    const int tm0 = blockIdx.y * BM;
    const int tn0 = blockIdx.x * BN;

    // cooperative global->LDS mapping: 16B (8-elem) chunks
    // A: 64 rows x 4 chunks = 256 chunks  -> 1 per thread
    // B: 128 rows x 4 chunks = 512 chunks -> 2 per thread
    const int crow = tid >> 2;            // 0..63
    const int ccol = (tid & 3) * 8;       // 0,8,16,24
    const __bf16* ag  = A + (size_t)(tm0 + crow) * K + ccol;
    const __bf16* bg0 = W + (size_t)(tn0 + crow) * K + ccol;
    const __bf16* bg1 = W + (size_t)(tn0 + crow + 64) * K + ccol;
    __bf16* asd  = As + crow * SA + ccol;
    __bf16* bsd0 = Bs + crow * SB + ccol;
    __bf16* bsd1 = Bs + (crow + 64) * SB + ccol;

    // fragment base pointers in LDS (ISA 7.12.2 layouts)
    // A 16x32: lane m=l16; elems 0-7 -> K=8*half+e, elems 8-15 -> K=16+8*half+e
    // B 32x16: lane n=l16; elems 0-7 -> K=8*half+e, elems 8-15 -> K=16+8*half+e
    const __bf16* ar0 = As + (wm * 32 + l16)      * SA + 8 * half;
    const __bf16* ar1 = As + (wm * 32 + 16 + l16) * SA + 8 * half;
    const __bf16* br0 = Bs + (wn * 32 + l16)      * SB + 8 * half;
    const __bf16* br1 = Bs + (wn * 32 + 16 + l16) * SB + 8 * half;

    v8f_t acc[2][2] = {};
    for (int k0 = 0; k0 < K; k0 += BK) {
        __syncthreads();
        *(v8bf_t*)asd  = *(const v8bf_t*)(ag  + k0);
        *(v8bf_t*)bsd0 = *(const v8bf_t*)(bg0 + k0);
        *(v8bf_t*)bsd1 = *(const v8bf_t*)(bg1 + k0);
        if (k0 + 4 * BK < K) {                       // global_prefetch_b8
            __builtin_prefetch(ag  + k0 + 4 * BK, 0, 0);
            __builtin_prefetch(bg0 + k0 + 4 * BK, 0, 0);
            __builtin_prefetch(bg1 + k0 + 4 * BK, 0, 0);
        }
        __syncthreads();

        FragBF a[2], b[2];
        a[0].h[0] = *(const v8bf_t*)(ar0);
        a[0].h[1] = *(const v8bf_t*)(ar0 + 16);
        a[1].h[0] = *(const v8bf_t*)(ar1);
        a[1].h[1] = *(const v8bf_t*)(ar1 + 16);
        b[0].h[0] = *(const v8bf_t*)(br0);
        b[0].h[1] = *(const v8bf_t*)(br0 + 16);
        b[1].h[0] = *(const v8bf_t*)(br1);
        b[1].h[1] = *(const v8bf_t*)(br1 + 16);

#pragma unroll
        for (int mt = 0; mt < 2; ++mt)
#pragma unroll
            for (int nt = 0; nt < 2; ++nt)
                acc[mt][nt] = __builtin_amdgcn_wmma_f32_16x16x32_bf16(
                    /*neg_a=*/false, a[mt].v, /*neg_b=*/false, b[nt].v,
                    /*c_mod=*/(short)0, acc[mt][nt],
                    /*reuse_a=*/false, /*reuse_b=*/false);
    }

    // D layout: lane col = l16; VGPR r holds row r + 8*half
#pragma unroll
    for (int mt = 0; mt < 2; ++mt) {
#pragma unroll
        for (int nt = 0; nt < 2; ++nt) {
            const int col = tn0 + wn * 32 + nt * 16 + l16;
            const float bv = bias[col];
#pragma unroll
            for (int r = 0; r < 8; ++r) {
                int row = tm0 + wm * 32 + mt * 16 + half * 8 + r;
                float v = acc[mt][nt][r] + bv;
                if (mode == 1) {
                    v = (v > 20.f) ? v : log1pf(__expf(v));    // softplus
                    v = fminf(fmaxf(v, 1e-4f), 10.f);          // clip
                }
                C[(size_t)row * N + col] = v;
            }
        }
    }
}

// --------------------------- causal depthwise conv + SiLU ------------------
__global__ __launch_bounds__(256) void conv_silu_kernel(
    const float* __restrict__ xz, const float* __restrict__ cw,
    const float* __restrict__ cb, float* __restrict__ xc,
    __bf16* __restrict__ xcb)
{
    int idx = blockIdx.x * 256 + threadIdx.x;
    if (idx >= Bb * Ll * DI) return;
    int i = idx % DI;
    int t = (idx / DI) % Ll;
    int b = idx / (DI * Ll);
    const float* w = cw + i * 4;
    float acc = cb[i];
#pragma unroll
    for (int j = 0; j < 4; ++j) {
        int t2 = t - 3 + j;                       // left pad d_conv-1 = 3
        if (t2 >= 0)
            acc += xz[(size_t)(b * Ll + t2) * (2 * DI) + i] * w[j];
    }
    float s = acc / (1.f + __expf(-acc));         // SiLU
    xc[idx]  = s;
    xcb[idx] = (__bf16)s;
}

// --------------------------- x_proj (N = 32, tiny) -------------------------
__global__ __launch_bounds__(32) void xproj_kernel(
    const float* __restrict__ xc, const float* __restrict__ W,
    const float* __restrict__ bias, float* __restrict__ BC)
{
    int m = blockIdx.x;
    int n = threadIdx.x;                           // one wave: 32 outputs
    const float* xr = xc + (size_t)m * DI;         // broadcast across lanes
    const float* wr = W  + (size_t)n * DI;
    float acc = bias[n];
    for (int k = 0; k < DI; k += 4) {
        acc += xr[k]   * wr[k];
        acc += xr[k+1] * wr[k+1];
        acc += xr[k+2] * wr[k+2];
        acc += xr[k+3] * wr[k+3];
    }
    BC[(size_t)m * 32 + n] = acc;
}

// --------------------------- selective scan --------------------------------
// one thread per (batch, channel); 16-wide state in registers
__global__ __launch_bounds__(256) void ssm_scan_kernel(
    const float* __restrict__ delta, const float* __restrict__ BC,
    const float* __restrict__ xc, const float* __restrict__ A_log,
    const float* __restrict__ Dvec, float* __restrict__ y)
{
    int tid = blockIdx.x * 256 + threadIdx.x;
    if (tid >= Bb * DI) return;
    int b = tid / DI;
    int i = tid % DI;

    float Arow[DS];
#pragma unroll
    for (int s = 0; s < DS; ++s) Arow[s] = -__expf(A_log[i * DS + s]);
    float Dv = Dvec[i];
    float h[DS];
#pragma unroll
    for (int s = 0; s < DS; ++s) h[s] = 0.f;

    for (int t = 0; t < Ll; ++t) {
        size_t m = (size_t)b * Ll + t;
        float d  = delta[m * DI + i];
        float xt = xc[m * DI + i];
        const float* bc = BC + m * 32;             // [0:16)=B_t, [16:32)=C_t
        float dx = d * xt;
#pragma unroll
        for (int s = 0; s < DS; ++s) {
            float da = d * Arow[s];
            da = fmaxf(da, -10.f);
            da = fminf(da, 0.f);
            h[s] = __expf(da) * h[s] + dx * bc[s];
        }
        if ((t & 15) == 15) {                      // periodic renorm
            float nn = 0.f;
#pragma unroll
            for (int s = 0; s < DS; ++s) nn += h[s] * h[s];
            float hn = fmaxf(sqrtf(nn), 1e-6f);
            float scale = (hn > 10.f) ? (10.f / hn) : 1.f;
#pragma unroll
            for (int s = 0; s < DS; ++s) h[s] *= scale;
        }
        float acc = 0.f;
#pragma unroll
        for (int s = 0; s < DS; ++s) acc += bc[16 + s] * h[s];
        y[m * DI + i] = acc + Dv * xt;
    }
}

// --------------------------- gate: y * silu(z) -> bf16 ---------------------
__global__ __launch_bounds__(256) void gate_kernel(
    const float* __restrict__ y, const float* __restrict__ xz,
    __bf16* __restrict__ gated)
{
    int idx = blockIdx.x * 256 + threadIdx.x;
    if (idx >= Mrows * DI) return;
    int m = idx / DI;
    int i = idx % DI;
    float z = xz[(size_t)m * (2 * DI) + DI + i];
    float g = y[idx] * (z / (1.f + __expf(-z)));
    gated[idx] = (__bf16)g;
}

// ---------------------------------------------------------------------------
extern "C" void kernel_launch(void* const* d_in, const int* in_sizes, int n_in,
                              void* d_out, int out_size, void* d_ws, size_t ws_size,
                              hipStream_t stream)
{
    (void)in_sizes; (void)n_in; (void)out_size;

    const float* x          = (const float*)d_in[0];
    const float* in_proj_w  = (const float*)d_in[1];
    const float* in_proj_b  = (const float*)d_in[2];
    const float* conv_w     = (const float*)d_in[3];
    const float* conv_b     = (const float*)d_in[4];
    const float* x_proj_w   = (const float*)d_in[5];
    const float* x_proj_b   = (const float*)d_in[6];
    const float* dt_proj_w  = (const float*)d_in[7];
    const float* dt_proj_b  = (const float*)d_in[8];
    const float* A_log      = (const float*)d_in[9];
    const float* Dvec       = (const float*)d_in[10];
    const float* out_proj_w = (const float*)d_in[11];
    const float* out_proj_b = (const float*)d_in[12];
    const float* in_na      = (const float*)d_in[13];
    const float* in_nb      = (const float*)d_in[14];
    const float* na         = (const float*)d_in[15];
    const float* nb         = (const float*)d_in[16];

    // ---------------- workspace layout (256B aligned) ----------------
    char*  ws  = (char*)d_ws;
    size_t off = 0;
    auto alloc = [&](size_t bytes) -> void* {
        void* p = ws + off;
        off += (bytes + 255) & ~(size_t)255;
        return p;
    };
    float*  xz     = (float*) alloc((size_t)Mrows * 2 * DI * 4);   // 32 MB
    float*  xc     = (float*) alloc((size_t)Mrows * DI * 4);       // 16 MB
    float*  delta  = (float*) alloc((size_t)Mrows * DI * 4);       // 16 MB (reused as out_proj result)
    float*  ybuf   = (float*) alloc((size_t)Mrows * DI * 4);       // 16 MB
    float*  BC     = (float*) alloc((size_t)Mrows * 32 * 4);       // 256 KB
    __bf16* xn_bf  = (__bf16*)alloc((size_t)Mrows * DM * 2);       // 4 MB
    __bf16* xc_bf  = (__bf16*)alloc((size_t)Mrows * DI * 2);       // 8 MB (reused as gated)
    __bf16* w_in   = (__bf16*)alloc((size_t)2 * DI * DM * 2);      // 8 MB
    __bf16* w_dt   = (__bf16*)alloc((size_t)DI * DI * 2);          // 8 MB
    __bf16* w_out  = (__bf16*)alloc((size_t)DM * DI * 2);          // 4 MB
    if (off > ws_size) return;                                     // workspace too small
    float*  outtmp = delta;   // delta dead after scan
    __bf16* gated  = xc_bf;   // xc_bf dead after dt_proj GEMM

    // 1) weights -> bf16 (once per launch, deterministic)
    {
        int n0 = 2 * DI * DM, n1 = DI * DI, n2 = DM * DI;
        cvt_bf16_kernel<<<(n0 + 255) / 256, 256, 0, stream>>>(in_proj_w,  w_in,  n0);
        cvt_bf16_kernel<<<(n1 + 255) / 256, 256, 0, stream>>>(dt_proj_w,  w_dt,  n1);
        cvt_bf16_kernel<<<(n2 + 255) / 256, 256, 0, stream>>>(out_proj_w, w_out, n2);
    }

    // 2) input LayerNorm -> bf16 activations
    layernorm_in_kernel<<<Mrows, 256, 0, stream>>>(x, in_na, in_nb, xn_bf);

    // 3) in_proj GEMM: (2048x1024) x (4096x1024)^T -> xz (2048x4096)
    gemm_wmma_bf16_kernel<<<dim3((2 * DI) / BN, Mrows / BM), 256, 0, stream>>>(
        xn_bf, w_in, in_proj_b, xz, Mrows, 2 * DI, DM, 0);

    // 4) causal depthwise conv + SiLU on x_in half of xz
    conv_silu_kernel<<<(Mrows * DI + 255) / 256, 256, 0, stream>>>(
        xz, conv_w, conv_b, xc, xc_bf);

    // 5) x_proj (B_t, C_t): 2048 rows x 32 outputs
    xproj_kernel<<<Mrows, 32, 0, stream>>>(xc, x_proj_w, x_proj_b, BC);

    // 6) dt_proj GEMM + softplus + clip -> delta
    gemm_wmma_bf16_kernel<<<dim3(DI / BN, Mrows / BM), 256, 0, stream>>>(
        xc_bf, w_dt, dt_proj_b, delta, Mrows, DI, DI, 1);

    // 7) selective scan
    ssm_scan_kernel<<<(Bb * DI + 255) / 256, 256, 0, stream>>>(
        delta, BC, xc, A_log, Dvec, ybuf);

    // 8) gate: y * silu(z) -> bf16
    gate_kernel<<<(Mrows * DI + 255) / 256, 256, 0, stream>>>(ybuf, xz, gated);

    // 9) out_proj GEMM -> outtmp (2048x1024)
    gemm_wmma_bf16_kernel<<<dim3(DM / BN, Mrows / BM), 256, 0, stream>>>(
        gated, w_out, out_proj_b, outtmp, Mrows, DM, DI, 0);

    // 10) residual + final LayerNorm -> d_out
    layernorm_out_kernel<<<Mrows, 256, 0, stream>>>(
        x, outtmp, na, nb, (float*)d_out);
}
